// TrajWorldTransformer_12498354831401
// MI455X (gfx1250) — matrix operand order repaired and checked by reading, not compiled, not run on hardware
//
#include <hip/hip_runtime.h>
#include <math.h>

// ---------------------------------------------------------------------------
// TrajWorldTransformer forward for gfx1250 (MI455X), wave32 + WMMA f16->f32.
// v2: f32->f16 conversion happens once at the global->LDS boundary; all LDS
// tiles are stored k-contiguous so every WMMA fragment is a contiguous 32B v16h load
// (the ISA A-operand K permutation is a swap of K-index bits 3 and 4).
// GEMM uses double-buffered f16 LDS tiles (one barrier per K step).
// ---------------------------------------------------------------------------

typedef __attribute__((ext_vector_type(16))) _Float16 v16h;
typedef __attribute__((ext_vector_type(4)))  _Float16 v4h;
typedef __attribute__((ext_vector_type(8)))  float    v8f;

#define NTOK  8192     // B*T*M
#define HDIM  512
#define FFDIM 2048
#define DINP  256
#define VOCSZ 256
#define TLEN  256
#define MVAR  16
#define NHEAD 8

static __device__ __forceinline__ v8f wmma_f16(v16h a, v16h b, v8f c) {
  // (neg_a, A, neg_b, B, c_mod, C, reuse_a, reuse_b)
  return __builtin_amdgcn_wmma_f32_16x16x32_f16(false, a, false, b, (short)0, c,
                                                false, false);
}

// ISA 16-bit A-matrix 16x32 K permutation == swap bits 3 and 4 of K index.
// Storing a 32-wide K chunk with this swap makes the per-lane A fragment
// (16 halves at [row][kg*16 .. kg*16+15]) a single contiguous 32B load.
static __device__ __forceinline__ int kswap(int c) {
  return (c & ~24) | ((c & 8) << 1) | ((c & 16) >> 1);
}

// Direct A-operand K index for global-memory gathers (no LDS staging).
static __device__ __forceinline__ int a_kidx(int h, int kg) {
  int v = h >> 1, pos = h & 1;
  return (v & 3) * 2 + pos + (v >> 2) * 16 + kg * 8;
}

static __device__ __forceinline__ float gelu_tanh(float x) {
  const float c0 = 0.7978845608028654f;  // sqrt(2/pi)
  return 0.5f * x * (1.f + tanhf(c0 * (x + 0.044715f * x * x * x)));
}

// ---------------------------------------------------------------------------
// Generic fused GEMM: C = [res +] act(A @ W + bias)
// A: MxK f32 row-major, W: KxN f32 row-major. M%128==0, K%32==0, N%64==0.
// Block = 256 threads = 8 waves; block tile 128x64; wave tile 16x64.
// LDS tiles are f16: A row-major ([m][k]), B transposed ([n][k]); double buffered.
// ---------------------------------------------------------------------------
__global__ __launch_bounds__(256) void k_gemm(
    const float* __restrict__ A, const float* __restrict__ W,
    const float* __restrict__ bias, const float* __restrict__ res,
    float* __restrict__ C, int M, int K, int N, int act_gelu) {
  __shared__ __align__(32) _Float16 sA[2][128 * 32];  // [row][k]
  __shared__ __align__(32) _Float16 sB[2][64 * 32];   // [col][k]
  const int tid  = threadIdx.x;
  const int wave = tid >> 5;
  const int lane = tid & 31;
  const int row0 = blockIdx.x * 128;
  const int col0 = blockIdx.y * 64;
  const int kg   = lane >> 4;

  v8f acc[4] = {v8f{}, v8f{}, v8f{}, v8f{}};

  auto stage = [&](int buf, int k0) {
#pragma unroll
    for (int i = 0; i < 4; ++i) {  // A tile 128x32: 1024 float4, 4/thread
      int idx = tid + i * 256;
      int r = idx >> 3, c = (idx & 7) << 2;
      float4 a4 = *(const float4*)&A[(size_t)(row0 + r) * K + (k0 + c)];
      v4h h4;
      h4.x = (_Float16)a4.x; h4.y = (_Float16)a4.y;
      h4.z = (_Float16)a4.z; h4.w = (_Float16)a4.w;
      *(v4h*)&sA[buf][r * 32 + c] = h4;  // contiguous k layout
    }
#pragma unroll
    for (int i = 0; i < 2; ++i) {  // B tile 32x64: 512 float4, transpose store
      int idx = tid + i * 256;
      int r = idx >> 4, c = (idx & 15) << 2;  // k row r, cols c..c+3
      float4 b4 = *(const float4*)&W[(size_t)(k0 + r) * N + (col0 + c)];
      sB[buf][(c + 0) * 32 + r] = (_Float16)b4.x;
      sB[buf][(c + 1) * 32 + r] = (_Float16)b4.y;
      sB[buf][(c + 2) * 32 + r] = (_Float16)b4.z;
      sB[buf][(c + 3) * 32 + r] = (_Float16)b4.w;
    }
  };

  const int nk = K >> 5;
  stage(0, 0);
  for (int t = 0; t < nk; ++t) {
    __syncthreads();
    const int cur = t & 1;
    if (t + 2 < nk) {  // global_prefetch_b8 two tiles ahead
      __builtin_prefetch(&A[(size_t)(row0 + (tid >> 1)) * K + (t + 2) * 32], 0, 1);
      __builtin_prefetch(&W[(size_t)((t + 2) * 32 + (tid >> 3)) * N + col0], 0, 1);
    }
    if (t + 1 < nk) stage(cur ^ 1, (t + 1) << 5);
    v16h af = *(const v16h*)&sA[cur][(wave * 16 + (lane & 15)) * 32 + kg * 16];
#pragma unroll
    for (int nt = 0; nt < 4; ++nt) {
      v16h bf = *(const v16h*)&sB[cur][(nt * 16 + (lane & 15)) * 32 + kg * 16];
      acc[nt] = wmma_f16(af, bf, acc[nt]);
    }
  }

  // Epilogue. C/D layout: lane group cg, col = lane&15, row = i + 8*cg.
  const int ccol = lane & 15;
  const int cg   = lane >> 4;
#pragma unroll
  for (int nt = 0; nt < 4; ++nt) {
    int gc = col0 + nt * 16 + ccol;
    float bv = bias[gc];
#pragma unroll
    for (int i = 0; i < 8; ++i) {
      int gr = row0 + wave * 16 + i + 8 * cg;
      float val = acc[nt][i] + bv;
      if (act_gelu) val = gelu_tanh(val);
      if (res) val += res[(size_t)gr * N + gc];
      C[(size_t)gr * N + gc] = val;
    }
  }
}

// ---------------------------------------------------------------------------
// x[n,d] += embed_obs_act[ind[n],d] + embed_timestep[t,d] + embed_variate[m,d]
// ---------------------------------------------------------------------------
__global__ __launch_bounds__(256) void k_embed_add(
    float* __restrict__ x, const int* __restrict__ ind,
    const float* __restrict__ eo, const float* __restrict__ et,
    const float* __restrict__ ev) {
  int idx = blockIdx.x * 256 + threadIdx.x;  // over NTOK*HDIM
  int n = idx >> 9;
  int d = idx & 511;
  int m = n & 15;
  int t = (n >> 4) & 255;
  x[idx] += eo[ind[n] * HDIM + d] + et[t * HDIM + d] + ev[m * HDIM + d];
}

// ---------------------------------------------------------------------------
// In-place LayerNorm over last dim (512). One block of 128 threads per row.
// ---------------------------------------------------------------------------
__global__ __launch_bounds__(128) void k_ln(float* __restrict__ x,
                                            const float* __restrict__ scale,
                                            const float* __restrict__ bias) {
  __shared__ float red[128];
  const int tid = threadIdx.x;
  const size_t base = (size_t)blockIdx.x * HDIM;
  float v[4];
  float s = 0.f;
#pragma unroll
  for (int i = 0; i < 4; ++i) {
    v[i] = x[base + tid + i * 128];
    s += v[i];
  }
  red[tid] = s;
  __syncthreads();
  for (int st = 64; st > 0; st >>= 1) {
    if (tid < st) red[tid] += red[tid + st];
    __syncthreads();
  }
  float mu = red[0] * (1.f / HDIM);
  __syncthreads();
  float s2 = 0.f;
#pragma unroll
  for (int i = 0; i < 4; ++i) {
    float dd = v[i] - mu;
    s2 += dd * dd;
  }
  red[tid] = s2;
  __syncthreads();
  for (int st = 64; st > 0; st >>= 1) {
    if (tid < st) red[tid] += red[tid + st];
    __syncthreads();
  }
  float inv = rsqrtf(red[0] * (1.f / HDIM) + 1e-6f);
#pragma unroll
  for (int i = 0; i < 4; ++i) {
    int c = tid + i * 128;
    x[base + c] = (v[i] - mu) * inv * scale[c] + bias[c];
  }
}

// ---------------------------------------------------------------------------
// Temporal causal attention. One block per (b, m, head): T=256, Dh=64.
// LDS (273KB dynamic): K f16 [f][d], V f16 transposed [d][f], per-wave q tile
// (f16, row-major), per-wave f32 score tile, per-wave f16 row-major P tile.
// Every WMMA fragment in the inner loops is one contiguous v16h LDS load.
// Token n(b,t,m) = ((b*256 + t)*16 + m); head slice = cols [h*64, h*64+64).
// ---------------------------------------------------------------------------
#define ATTN_T_LDS (256*64*2 + 64*256*2 + 8*16*64*2 + 8*16*256*2 + 256*4 + 8*16*256*4)

__global__ __launch_bounds__(256) void k_attn_temporal(
    const float* __restrict__ q, const float* __restrict__ k,
    const float* __restrict__ v, const float* __restrict__ pmask,
    float* __restrict__ out) {
  extern __shared__ __align__(32) char smem[];
  _Float16* kf  = (_Float16*)smem;               // [256][64]
  _Float16* vt  = kf + 256 * 64;                 // [64][256] (transposed)
  _Float16* qb  = vt + 64 * 256;                 // [8][16][64]  row-major
  _Float16* p16 = qb + 8 * 16 * 64;              // [8][16][256] row-major
  float* pm     = (float*)(p16 + 8 * 16 * 256);  // [256]
  float* scAll  = pm + 256;                      // [8][16][256] f32
  const int tid = threadIdx.x, wave = tid >> 5, lane = tid & 31;
  const int h  = blockIdx.x & 7;
  const int bm = blockIdx.x >> 3;
  const int m  = bm & 15;
  const int b  = bm >> 4;
  _Float16* qw = qb + wave * (16 * 64);
  _Float16* pw = p16 + wave * (16 * 256);
  float* sc    = scAll + wave * (16 * 256);

  {  // stage K (row-major) and V (transposed), f32 -> f16; one key row/thread
    const int f = tid;
    size_t base = ((size_t)(b * 256 + f) * 16 + m) * 512 + h * 64;
#pragma unroll
    for (int d = 0; d < 64; d += 4) {
      float4 k4 = *(const float4*)&k[base + d];
      float4 v4 = *(const float4*)&v[base + d];
      v4h h4;
      h4.x = (_Float16)k4.x; h4.y = (_Float16)k4.y;
      h4.z = (_Float16)k4.z; h4.w = (_Float16)k4.w;
      *(v4h*)&kf[f * 64 + d] = h4;
      vt[(d + 0) * 256 + f] = (_Float16)v4.x;
      vt[(d + 1) * 256 + f] = (_Float16)v4.y;
      vt[(d + 2) * 256 + f] = (_Float16)v4.z;
      vt[(d + 3) * 256 + f] = (_Float16)v4.w;
    }
    pm[f] = pmask[b * 256 + f];
  }
  __syncthreads();

  const int ccol = lane & 15;
  const int cg   = lane >> 4;
  const int kg   = lane >> 4;

  // each wave owns q-row-tiles {wave, wave+8}: uniform trip count => barriers OK
  for (int qt = wave; qt < 16; qt += 8) {
    const int q0 = qt * 16;
    {  // stage this wave's 16x64 q tile, f16, row-major
      const int r = lane & 15, hl = lane >> 4;
      size_t qrow =
          ((size_t)(b * 256 + q0 + r) * 16 + m) * 512 + h * 64 + hl * 32;
#pragma unroll
      for (int j = 0; j < 32; j += 4) {
        float4 q4 = *(const float4*)&q[qrow + j];
        v4h h4;
        h4.x = (_Float16)q4.x; h4.y = (_Float16)q4.y;
        h4.z = (_Float16)q4.z; h4.w = (_Float16)q4.w;
        *(v4h*)&qw[r * 64 + hl * 32 + j] = h4;
      }
    }
    __syncthreads();
    v16h aq0 = *(const v16h*)&qw[(lane & 15) * 64 + kg * 16];
    v16h aq1 = *(const v16h*)&qw[(lane & 15) * 64 + 32 + kg * 16];
    // scores = q @ k^T  (B operand: kf[f][d] is contiguous in d)
    for (int ft = 0; ft < 16; ++ft) {
      const int col = ft * 16 + ccol;
      v8f d = {};
      v16h bf0 = *(const v16h*)&kf[col * 64 + kg * 16];
      d = wmma_f16(aq0, bf0, d);
      v16h bf1 = *(const v16h*)&kf[col * 64 + 32 + kg * 16];
      d = wmma_f16(aq1, bf1, d);
#pragma unroll
      for (int i = 0; i < 8; ++i) sc[(i + 8 * cg) * 256 + col] = d[i];
    }
    __syncthreads();
    {  // scale, causal+pad mask, softmax; emit normalized P as row-major f16
      const int r = lane & 15, hl = lane >> 4;
      const int qrow = q0 + r;
      const int f0 = hl * 128;
      float mx = -1e30f;
      for (int f = f0; f < f0 + 128; ++f) {
        float wv = sc[r * 256 + f] * 0.125f;  // 1/sqrt(64)
        if (f > qrow) wv = -1e30f;            // causal -inf
        if (pm[f] == 0.f) wv = -10000.f;      // pmask overrides
        sc[r * 256 + f] = wv;
        mx = fmaxf(mx, wv);
      }
      mx = fmaxf(mx, __shfl_xor(mx, 16, 32));
      float sum = 0.f;
      for (int f = f0; f < f0 + 128; ++f) {
        float e = __expf(sc[r * 256 + f] - mx);
        sc[r * 256 + f] = e;
        sum += e;
      }
      sum += __shfl_xor(sum, 16, 32);
      float inv = 1.f / sum;
      for (int f = f0; f < f0 + 128; ++f)
        pw[r * 256 + f] =
            (_Float16)(sc[r * 256 + f] * inv);
    }
    __syncthreads();
    // out = P @ v  (A = row-major P, B = vt[d][f]: both contiguous v16h loads)
    v8f oacc[4] = {v8f{}, v8f{}, v8f{}, v8f{}};
    for (int kt = 0; kt < 8; ++kt) {
      v16h ap = *(const v16h*)&pw[(lane & 15) * 256 + kt * 32 + kg * 16];
#pragma unroll
      for (int dt = 0; dt < 4; ++dt) {
        v16h bvf =
            *(const v16h*)&vt[(dt * 16 + ccol) * 256 + kt * 32 + kg * 16];
        oacc[dt] = wmma_f16(ap, bvf, oacc[dt]);
      }
    }
#pragma unroll
    for (int dt = 0; dt < 4; ++dt)
#pragma unroll
      for (int i = 0; i < 8; ++i) {
        size_t on = ((size_t)(b * 256 + q0 + i + 8 * cg) * 16 + m) * 512 +
                    h * 64 + dt * 16 + ccol;
        out[on] = oacc[dt][i];
      }
    __syncthreads();
  }
}

// ---------------------------------------------------------------------------
// Variate (non-causal) attention: seq len M=16. One wave per (b*T + t, head).
// Scores in one WMMA accumulator; softmax via cross-lane shuffles; normalized
// P written as row-major f16 into a zero-padded (K=32) per-wave LDS tile so
// the P@V A-fragment is one contiguous v16h load.
// ---------------------------------------------------------------------------
__global__ __launch_bounds__(256) void k_attn_variate(
    const float* __restrict__ q, const float* __restrict__ k,
    const float* __restrict__ v, float* __restrict__ out) {
  __shared__ __align__(32) _Float16 sp[8][16 * 32];
  const int tid = threadIdx.x, wave = tid >> 5, lane = tid & 31;
  const int gw = blockIdx.x * 8 + wave;  // (bt, head)
  const int h  = gw & 7;
  const int bt = gw >> 3;
  const size_t base = (size_t)bt * 16 * 512 + h * 64;  // + m*512 + d
  const int arow = lane & 15, kg = lane >> 4;
  const int ccol = lane & 15, cg = lane >> 4;

  {  // zero-fill P tile (covers the K=16..31 padding rows)
    v4h z = {};
#pragma unroll
    for (int j = 0; j < 16; j += 4) *(v4h*)&sp[wave][lane * 16 + j] = z;
  }

  v8f d = {};
#pragma unroll
  for (int c = 0; c < 2; ++c) {  // K = 64 over two 32-chunks
    v16h aq, bk;
#pragma unroll
    for (int hh = 0; hh < 16; ++hh)
      aq[hh] = (_Float16)q[base + (size_t)arow * 512 + c * 32 + kg * 16 + hh];
    const int kb = kg * 16 + c * 32;
#pragma unroll
    for (int hh = 0; hh < 16; ++hh)
      bk[hh] = (_Float16)k[base + (size_t)ccol * 512 + kb + hh];
    d = wmma_f16(aq, bk, d);
  }
  // softmax over N (cols of one row live across the 16 lanes of a group)
  float p8[8];
#pragma unroll
  for (int i = 0; i < 8; ++i) {
    float wv = d[i] * 0.125f;
    float mx = wv;
#pragma unroll
    for (int off = 1; off < 16; off <<= 1)
      mx = fmaxf(mx, __shfl_xor(mx, off, 32));
    float e = __expf(wv - mx);
    float s = e;
#pragma unroll
    for (int off = 1; off < 16; off <<= 1) s += __shfl_xor(s, off, 32);
    p8[i] = e / s;
  }
#pragma unroll
  for (int i = 0; i < 8; ++i)
    sp[wave][(i + 8 * cg) * 32 + ccol] = (_Float16)p8[i];
  __syncthreads();

  v16h ap = *(const v16h*)&sp[wave][arow * 32 + kg * 16];
#pragma unroll
  for (int dt = 0; dt < 4; ++dt) {
    v16h bvf;
    const int kb = kg * 16;
#pragma unroll
    for (int hh = 0; hh < 16; ++hh) {
      int kk = kb + hh;
      bvf[hh] = (kk < 16)
                    ? (_Float16)v[base + (size_t)kk * 512 + dt * 16 + ccol]
                    : (_Float16)0.f;
    }
    v8f o = {};
    o = wmma_f16(ap, bvf, o);
#pragma unroll
    for (int i = 0; i < 8; ++i)
      out[base + (size_t)(i + 8 * cg) * 512 + dt * 16 + ccol] = o[i];
  }
}

// ---------------------------------------------------------------------------
// Host orchestration
// ---------------------------------------------------------------------------
static inline void launch_gemm(const float* A, const float* W, const float* b,
                               const float* res, float* C, int M, int K, int N,
                               int act, hipStream_t s) {
  dim3 grid(M / 128, N / 64);
  k_gemm<<<grid, dim3(256), 0, s>>>(A, W, b, res, C, M, K, N, act);
}

extern "C" void kernel_launch(void* const* d_in, const int* in_sizes, int n_in,
                              void* d_out, int out_size, void* d_ws,
                              size_t ws_size, hipStream_t stream) {
  (void)in_sizes; (void)n_in; (void)out_size; (void)ws_size;
  // JAX pytree (alphabetical) flattening:
  // 0 inputs, 1 obs_act_indicator, 2 padding_mask,
  // blocks i at base=3+32*i:
  //   +0..7  attn0: k.b,k.w,o.b,o.w,q.b,q.w,v.b,v.w
  //   +8..15 attn1: same order
  //   +16/17 ln0 bias/scale, +18/19 ln1, +20/21 ln2, +22/23 ln3
  //   +24/25 mlp0_fc b/w, +26/27 mlp0_proj b/w, +28/29 mlp1_fc, +30/31 mlp1_proj
  // 131 embed_obs_act, 132/133 embed_proj b/w, 134 embed_timestep,
  // 135 embed_variate, 136/137 head b/w
  const float* inputs = (const float*)d_in[0];
  const int*   ind    = (const int*)d_in[1];
  const float* pmask  = (const float*)d_in[2];
  auto P = [&](int i) { return (const float*)d_in[i]; };

  float* ws = (float*)d_ws;
  const size_t S = (size_t)NTOK * HDIM;
  float* X  = ws;          // activations (8192 x 512)
  float* Q  = ws + 1 * S;
  float* Kb = ws + 2 * S;
  float* Vb = ws + 3 * S;
  float* AT = ws + 4 * S;  // attention output
  float* MB = ws + 5 * S;  // MLP hidden (8192 x 2048) = 4*S floats

  hipFuncSetAttribute((const void*)k_attn_temporal,
                      hipFuncAttributeMaxDynamicSharedMemorySize, ATTN_T_LDS);

  // embed: X = inputs @ embed_proj.w + b, then += embedding tables
  launch_gemm(inputs, P(133), P(132), nullptr, X, NTOK, DINP, HDIM, 0, stream);
  k_embed_add<<<NTOK * HDIM / 256, 256, 0, stream>>>(X, ind, P(131), P(134),
                                                     P(135));

  for (int blk = 0; blk < 4; ++blk) {
    const int bb = 3 + 32 * blk;
    // --- temporal attention (attn1, causal) ---
    launch_gemm(X, P(bb + 13), P(bb + 12), nullptr, Q,  NTOK, HDIM, HDIM, 0, stream); // q
    launch_gemm(X, P(bb + 9),  P(bb + 8),  nullptr, Kb, NTOK, HDIM, HDIM, 0, stream); // k
    launch_gemm(X, P(bb + 15), P(bb + 14), nullptr, Vb, NTOK, HDIM, HDIM, 0, stream); // v
    k_attn_temporal<<<2 * MVAR * NHEAD, 256, ATTN_T_LDS, stream>>>(Q, Kb, Vb,
                                                                   pmask, AT);
    launch_gemm(AT, P(bb + 11), P(bb + 10), X, X, NTOK, HDIM, HDIM, 0, stream); // o + res
    k_ln<<<NTOK, 128, 0, stream>>>(X, P(bb + 19), P(bb + 18));                  // ln1
    launch_gemm(X,  P(bb + 29), P(bb + 28), nullptr, MB, NTOK, HDIM, FFDIM, 1, stream); // mlp1_fc+gelu
    launch_gemm(MB, P(bb + 31), P(bb + 30), X, X, NTOK, FFDIM, HDIM, 0, stream);        // mlp1_proj+res
    k_ln<<<NTOK, 128, 0, stream>>>(X, P(bb + 23), P(bb + 22));                  // ln3
    // --- variate attention (attn0, non-causal) ---
    launch_gemm(X, P(bb + 5), P(bb + 4), nullptr, Q,  NTOK, HDIM, HDIM, 0, stream); // q
    launch_gemm(X, P(bb + 1), P(bb + 0), nullptr, Kb, NTOK, HDIM, HDIM, 0, stream); // k
    launch_gemm(X, P(bb + 7), P(bb + 6), nullptr, Vb, NTOK, HDIM, HDIM, 0, stream); // v
    k_attn_variate<<<2 * TLEN * NHEAD / 8, 256, 0, stream>>>(Q, Kb, Vb, AT);
    launch_gemm(AT, P(bb + 3), P(bb + 2), X, X, NTOK, HDIM, HDIM, 0, stream);   // o + res
    k_ln<<<NTOK, 128, 0, stream>>>(X, P(bb + 17), P(bb + 16));                  // ln0
    launch_gemm(X,  P(bb + 25), P(bb + 24), nullptr, MB, NTOK, HDIM, FFDIM, 1, stream); // mlp0_fc+gelu
    launch_gemm(MB, P(bb + 27), P(bb + 26), X, X, NTOK, FFDIM, HDIM, 0, stream);        // mlp0_proj+res
    k_ln<<<NTOK, 128, 0, stream>>>(X, P(bb + 21), P(bb + 20));                  // ln2
  }

  // head: d_out = X @ head.w + head.b   (8192 x 256)
  launch_gemm(X, P(137), P(136), nullptr, (float*)d_out, NTOK, HDIM, VOCSZ, 0,
              stream);
}
